// StructureUpdate_22204980920955
// MI455X (gfx1250) — matrix operand order repaired and checked by reading, not compiled, not run on hardware
//
#include <hip/hip_runtime.h>

// ---------------------------------------------------------------------------
// IPA (Invariant Point Attention) block + MLP for MI455X (gfx1250, wave32).
// All GEMM-shaped math uses v_wmma_f32_16x16x32_f16 (f16 in, f32 accumulate).
// Layout assumptions (CDNA5 ISA 7.12.2):
//   A (16x32 f16): lane l -> row M=l%16, half=l/16; element e -> K =
//       half*16 + e   (contiguous, chosen to match B; not the ISA table order)
//   B (32x16 f16): lane l -> col N=l%16, half=l/16; element e -> K = half*16 + e (chosen here, same as A; ISA table order differs)
//   C/D (16x16 f32): lane l -> col N=l%16; VGPR v -> row M = v + 8*(l/16)
// LDS B-tiles are staged n-major with a 40-f16 (80 B) pitch so each lane's
// fragment is two aligned 16B runs (-> ds_load_b128 x2) and staging writes
// land in distinct banks (20 dwords/row).
// ---------------------------------------------------------------------------

#define B_   2
#define N_   512
#define D_   384
#define E_   128
#define H_   8
#define SK_  16
#define SV_  16
#define PK_  4
#define PV_  4
#define DH_  768
#define BN_  (B_*N_)
#define QKV_ 672      // 3*H*SK(=128*3) + 3*H*PK*3(=96*3)
#define CAT_ 1280     // H*(SV + PV*3 + PV + E)
#define BP_  40       // padded B-tile pitch in f16 (80 bytes)

typedef __attribute__((ext_vector_type(16))) _Float16 v16h;
typedef __attribute__((ext_vector_type(4)))  _Float16 v4h;
typedef __attribute__((ext_vector_type(8)))  float    v8f;

__device__ __forceinline__ int kmapA(int e, int half) {
  return (half << 4) + e;
}

// ---------------------------------------------------------------------------
// f32 -> f16 conversion
// ---------------------------------------------------------------------------
__global__ void cvt_kernel(const float* __restrict__ s, _Float16* __restrict__ d, int n) {
  int i = blockIdx.x * blockDim.x + threadIdx.x;
  if (i < n) d[i] = (_Float16)s[i];
}

// Pack [Wq|Wk|Wv|Wpq|Wpk|Wpv] -> (D, 672) f16
__global__ void pack_wcat_kernel(const float* __restrict__ Wq, const float* __restrict__ Wk,
                                 const float* __restrict__ Wv, const float* __restrict__ Wpq,
                                 const float* __restrict__ Wpk, const float* __restrict__ Wpv,
                                 _Float16* __restrict__ W, int n) {
  int i = blockIdx.x * blockDim.x + threadIdx.x;
  if (i >= n) return;
  int k = i / QKV_, c = i % QKV_;
  float v;
  if      (c < 128) v = Wq [k*128 + c];
  else if (c < 256) v = Wk [k*128 + (c-128)];
  else if (c < 384) v = Wv [k*128 + (c-256)];
  else if (c < 480) v = Wpq[k*96  + (c-384)];
  else if (c < 576) v = Wpk[k*96  + (c-480)];
  else              v = Wpv[k*96  + (c-576)];
  W[i] = (_Float16)v;
}

// ---------------------------------------------------------------------------
// Generic WMMA GEMM: C[M,N] = relu?(A[M,K] @ B[K,N] + bias)
// block = 128 threads (4 waves); wave w owns one 16x16 N-tile; A staged once
// per block per K-step, B staged per wave (n-major, padded pitch).
// ---------------------------------------------------------------------------
__global__ void __launch_bounds__(128) gemm_f16_kernel(
    const _Float16* __restrict__ A, const _Float16* __restrict__ Bm,
    const float* __restrict__ bias, float* __restrict__ C,
    int M, int N, int K, int relu)
{
  __shared__ _Float16 sA[16 * 32];
  __shared__ _Float16 sB[4][16 * BP_];
  const int tid  = threadIdx.x;
  const int wave = tid >> 5, lane = tid & 31;
  const int half = lane >> 4, lm = lane & 15;
  const int ntiles = N >> 4;
  const int nt = blockIdx.x * 4 + wave;
  const int m0 = blockIdx.y << 4;
  const int n0 = nt << 4;
  const bool active = (nt < ntiles);
  v8f acc = {};
  const int ksteps = K >> 5;
  for (int ks = 0; ks < ksteps; ++ks) {
    const int k0 = ks << 5;
    for (int r = 0; r < 4; ++r) {                 // stage A (16x32, k-major)
      int idx = tid + (r << 7);
      int mm = idx >> 5, kk = idx & 31;
      sA[idx] = A[(size_t)(m0 + mm) * K + k0 + kk];
    }
    if (active) {                                 // stage B (n-major, pitch 40)
      for (int r = 0; r < 16; ++r) {
        int idx = (r << 5) + lane;
        int kk = idx >> 4, nn = idx & 15;
        sB[wave][nn * BP_ + kk] = Bm[(size_t)(k0 + kk) * N + n0 + nn];
      }
    }
    __syncthreads();
    if (active) {
      v16h af, bf;
      #pragma unroll
      for (int e = 0; e < 16; ++e) {
        af[e] = sA[lm * 32 + kmapA(e, half)];
        bf[e] = sB[wave][lm * BP_ + (half << 4) + e];   // contiguous 16B runs
      }
      acc = __builtin_amdgcn_wmma_f32_16x16x32_f16(false, af, false, bf,
                                                   (short)0, acc, false, false);
    }
    __syncthreads();
  }
  if (active) {
    const int ng = n0 + lm;
    #pragma unroll
    for (int v = 0; v < 8; ++v) {
      int mg = m0 + v + (half << 3);
      float val = acc[v];
      if (bias) val += bias[ng];
      if (relu) val = fmaxf(val, 0.f);
      C[(size_t)mg * N + ng] = val;
    }
  }
}

// ---------------------------------------------------------------------------
// Apply frames: qp/kp/vp[r] = sum_c proj[c] * rot[b,n,c,r] + trans[r]
// raw cols 384..480 = pq, 480..576 = pk, 576..672 = pv
// ---------------------------------------------------------------------------
__global__ void rotpts_kernel(const float* __restrict__ raw, const float* __restrict__ rot,
                              const float* __restrict__ trans,
                              float* __restrict__ qp, float* __restrict__ kp,
                              float* __restrict__ vp, int n)
{
  int i = blockIdx.x * blockDim.x + threadIdx.x;
  if (i >= n) return;
  int row = i / 96, q = i % 96;
  int r = q % 3, base = q - r;                 // h*12 + p*3
  const float* rw = raw + (size_t)row * QKV_ + 384;
  const float* R  = rot + (size_t)row * 9;
  float t = trans[row * 3 + r];
  float a = 0.f, b = 0.f, c = 0.f;
  #pragma unroll
  for (int cc = 0; cc < 3; ++cc) {
    float rr = R[cc * 3 + r];
    a += rw[base + cc]       * rr;
    b += rw[96  + base + cc] * rr;
    c += rw[192 + base + cc] * rr;
  }
  qp[i] = a + t; kp[i] = b + t; vp[i] = c + t;
}

// ---------------------------------------------------------------------------
// Fused attention: one workgroup (8 waves) per (b,i).
//   1a: scalar qk dot + point distance logits (VALU)
//   1b: pair bias edge@Wpb via WMMA (fused, edge streamed f32->f16 in LDS)
//   2 : per-head softmax (wave32 shuffles)
//   3a: res_e = attn@edge via WMMA (M=16 heads-padded, N=128, K=512)
//   3b: res_s/res_p (small) on VALU, inverse-frame rotation, point norms
// Mask is all-true in this problem -> masking elided.
// ---------------------------------------------------------------------------
__global__ void __launch_bounds__(256) attn_kernel(
    const float* __restrict__ raw,      // (BN,672)  qs|ks|vs|pq|pk|pv
    const float* __restrict__ qp, const float* __restrict__ kp,
    const float* __restrict__ vp,       // (BN,96) each, frame-applied
    const float* __restrict__ edge,     // (B,N,N,E) f32
    const _Float16* __restrict__ Wpbh,  // (E,H) f16
    const float* __restrict__ pt_w, const float* __restrict__ bpb,
    const float* __restrict__ rot, const float* __restrict__ trans,
    float* __restrict__ cat)            // (BN,1280)
{
  __shared__ float    att[H_][N_];      // 16 KB logits -> probs
  __shared__ _Float16 attnh[H_][N_];    //  8 KB f16 probs (WMMA A)
  __shared__ _Float16 stg[8][16 * BP_]; // 10 KB per-wave edge staging
  __shared__ float qrow[128], qprow[96], pws[H_], rtmp[96], rout[96];

  const int i = blockIdx.x, b = blockIdx.y;
  const int row = b * N_ + i;
  const int tid = threadIdx.x;
  const int wave = tid >> 5, lane = tid & 31;
  const int half = lane >> 4, lm = lane & 15;
  const float scalar_scale = rsqrtf(3.f * (float)SK_);
  const float point_scale  = rsqrtf(3.f * (float)PK_ * 4.5f);
  const float pair_scale   = rsqrtf(3.f);

  if (tid < 128) qrow[tid]  = raw[(size_t)row * QKV_ + tid];
  if (tid < 96)  qprow[tid] = qp[(size_t)row * 96 + tid];
  if (tid < H_)  pws[tid]   = log1pf(expf(pt_w[tid]));   // softplus
  __syncthreads();

  // ---- phase 1a: scalar + point logits -------------------------------------
  for (int r = 0; r < 16; ++r) {
    int idx = tid + (r << 8);                 // 0..4095 = (h,j)
    int h = idx >> 9, j = idx & 511;
    const float* kr = raw + (size_t)(b * N_ + j) * QKV_ + 128 + h * 16;
    float dot = 0.f;
    #pragma unroll
    for (int d = 0; d < 16; ++d) dot += qrow[h * 16 + d] * kr[d];
    const float* kpr = kp + (size_t)(b * N_ + j) * 96 + h * 12;
    float dist = 0.f;
    #pragma unroll
    for (int c = 0; c < 12; ++c) { float df = qprow[h * 12 + c] - kpr[c]; dist += df * df; }
    att[h][j] = dot * scalar_scale - 0.5f * point_scale * pws[h] * dist;
  }
  __syncthreads();

  // ---- phase 1b: pair bias via WMMA (A = Wpb^T padded to 16 rows) ----------
  // Hoist the four loop-invariant Wpb A-fragments (K = 0..127).
  v16h af_pb[4];
  #pragma unroll
  for (int ksd = 0; ksd < 4; ++ksd) {
    #pragma unroll
    for (int e = 0; e < 16; ++e) {
      int kA = (ksd << 5) + kmapA(e, half);
      af_pb[ksd][e] = (lm < H_) ? Wpbh[(size_t)kA * H_ + lm] : (_Float16)0.f;
    }
  }
  for (int jt = wave; jt < 32; jt += 8) {
    v8f acc = {};
    for (int ksd = 0; ksd < 4; ++ksd) {       // E = 128 in chunks of 32
      int e0 = ksd << 5;
      const float* ep = edge + (((size_t)b * N_ + i) * N_ + jt * 16) * E_ + e0;
      // stage B tile n-major: stg[jl*32 + el]  (jl = n, el = k_local)
      for (int jl = 0; jl < 16; ++jl)
        stg[wave][jl * 32 + lane] = (_Float16)ep[(size_t)jl * E_ + lane];
      v16h bf;
      #pragma unroll
      for (int e = 0; e < 16; ++e)
        bf[e] = stg[wave][lm * 32 + (half << 4) + e];   // contiguous 16B runs
      acc = __builtin_amdgcn_wmma_f32_16x16x32_f16(false, af_pb[ksd], false, bf,
                                                   (short)0, acc, false, false);
    }
    if (lane < 16) {
      int j = jt * 16 + lane;
      #pragma unroll
      for (int v = 0; v < H_; ++v)
        att[v][j] += (acc[v] + bpb[v]) * pair_scale;
    }
  }
  __syncthreads();

  // ---- phase 2: softmax, wave h owns head h --------------------------------
  {
    int h = wave;
    float mx = -3.402823466e+38f;
    for (int r = 0; r < 16; ++r) mx = fmaxf(mx, att[h][lane + 32 * r]);
    #pragma unroll
    for (int o = 16; o > 0; o >>= 1) mx = fmaxf(mx, __shfl_xor(mx, o, 32));
    float sum = 0.f;
    for (int r = 0; r < 16; ++r) {
      int j = lane + 32 * r;
      float e = expf(att[h][j] - mx);
      att[h][j] = e; sum += e;
    }
    #pragma unroll
    for (int o = 16; o > 0; o >>= 1) sum += __shfl_xor(sum, o, 32);
    float inv = 1.f / sum;
    for (int r = 0; r < 16; ++r) {
      int j = lane + 32 * r;
      float a = att[h][j] * inv;
      att[h][j] = a; attnh[h][j] = (_Float16)a;
    }
  }
  __syncthreads();

  // ---- phase 3a: res_e = attn @ edge via WMMA, wave w owns e-tile w --------
  {
    const int et = wave;
    v8f acc = {};
    const int el = lane & 15, jh = lane >> 4;
    for (int ksj = 0; ksj < 16; ++ksj) {      // K = 512 in chunks of 32
      int j0 = ksj << 5;
      const float* ep = edge + (((size_t)b * N_ + i) * N_ + j0) * E_ + et * 16;
      __builtin_prefetch(ep + 32 * E_, 0, 1);
      // stage B tile n-major, padded pitch: stg[el*BP_ + jl]
      for (int r = 0; r < 4; ++r) {
        int jl0 = (jh << 4) + (r << 2);
        v4h pk;
        #pragma unroll
        for (int c = 0; c < 4; ++c)
          pk[c] = (_Float16)ep[(size_t)(jl0 + c) * E_ + el];
        *(v4h*)&stg[wave][el * BP_ + jl0] = pk;   // 8B aligned vector store
      }
      v16h af, bf;
      #pragma unroll
      for (int e = 0; e < 16; ++e) {
        int kA = j0 + kmapA(e, half);
        af[e] = (lm < H_) ? attnh[lm][kA] : (_Float16)0.f;
        bf[e] = stg[wave][lm * BP_ + (half << 4) + e];  // contiguous 16B runs
      }
      acc = __builtin_amdgcn_wmma_f32_16x16x32_f16(false, af, false, bf,
                                                   (short)0, acc, false, false);
    }
    if (lane < 16) {
      #pragma unroll
      for (int v = 0; v < H_; ++v)
        cat[(size_t)row * CAT_ + 256 + v * E_ + et * 16 + lane] = acc[v];
    }
  }

  // ---- phase 3b: res_s, res_p (+inverse frame), res_pn ---------------------
  if (tid < 128) {                            // (h,d) -> res_s
    int h = tid >> 4, d = tid & 15;
    const float* vcol = raw + (size_t)(b * N_) * QKV_ + 256 + h * 16 + d;
    float s = 0.f;
    for (int j = 0; j < N_; ++j) s += att[h][j] * vcol[(size_t)j * QKV_];
    cat[(size_t)row * CAT_ + h * 16 + d] = s;
  }
  if (tid < 96) {                             // (h,p,c) -> attn @ vp
    const float* vcol = vp + (size_t)(b * N_) * 96 + tid;
    int h = tid / 12;
    float s = 0.f;
    for (int j = 0; j < N_; ++j) s += att[h][j] * vcol[(size_t)j * 96];
    rtmp[tid] = s;
  }
  __syncthreads();
  if (tid < 96) {                             // inverse frame: sum_c (x-t)[c]*R[r,c]
    int h = tid / 12, p = (tid % 12) / 3, r = tid % 3;
    float s = 0.f;
    #pragma unroll
    for (int c = 0; c < 3; ++c)
      s += (rtmp[h * 12 + p * 3 + c] - trans[row * 3 + c]) * rot[(size_t)row * 9 + r * 3 + c];
    rout[tid] = s;
    cat[(size_t)row * CAT_ + 128 + tid] = s;
  }
  __syncthreads();
  if (tid < 32) {                             // point norms
    int h = tid >> 2, p = tid & 3;
    float s = 1e-8f;
    #pragma unroll
    for (int r = 0; r < 3; ++r) { float v = rout[h * 12 + p * 3 + r]; s += v * v; }
    cat[(size_t)row * CAT_ + 224 + tid] = sqrtf(s);
  }
}

// ---------------------------------------------------------------------------
// LayerNorm: one 128-thread block per row; optional f32 and/or f16 outputs.
// ---------------------------------------------------------------------------
__global__ void __launch_bounds__(128) ln_kernel(
    const float* __restrict__ x, const float* __restrict__ g,
    const float* __restrict__ bta, float* __restrict__ outf,
    _Float16* __restrict__ outh, int D)
{
  __shared__ float red[4];
  int row = blockIdx.x, tid = threadIdx.x;
  const float* xr = x + (size_t)row * D;
  float s = 0.f;
  for (int c = tid; c < D; c += 128) s += xr[c];
  #pragma unroll
  for (int o = 16; o > 0; o >>= 1) s += __shfl_xor(s, o, 32);
  if ((tid & 31) == 0) red[tid >> 5] = s;
  __syncthreads();
  float mu = (red[0] + red[1] + red[2] + red[3]) / (float)D;
  __syncthreads();
  float v = 0.f;
  for (int c = tid; c < D; c += 128) { float d = xr[c] - mu; v += d * d; }
  #pragma unroll
  for (int o = 16; o > 0; o >>= 1) v += __shfl_xor(v, o, 32);
  if ((tid & 31) == 0) red[tid >> 5] = v;
  __syncthreads();
  float inv = rsqrtf((red[0] + red[1] + red[2] + red[3]) / (float)D + 1e-5f);
  for (int c = tid; c < D; c += 128) {
    float y = (xr[c] - mu) * inv * g[c] + bta[c];
    if (outf) outf[(size_t)row * D + c] = y;
    if (outh) outh[(size_t)row * D + c] = (_Float16)y;
  }
}

// ---------------------------------------------------------------------------
extern "C" void kernel_launch(void* const* d_in, const int* in_sizes, int n_in,
                              void* d_out, int out_size, void* d_ws, size_t ws_size,
                              hipStream_t stream) {
  (void)in_sizes; (void)n_in; (void)out_size; (void)ws_size;
  const float* node  = (const float*)d_in[0];
  const float* edge  = (const float*)d_in[1];
  const float* rot   = (const float*)d_in[2];
  const float* trans = (const float*)d_in[3];
  // d_in[4] = mask: all-true in this problem; masking is a no-op.
  const float* Wq  = (const float*)d_in[5];
  const float* Wk  = (const float*)d_in[6];
  const float* Wv  = (const float*)d_in[7];
  const float* Wpq = (const float*)d_in[8];
  const float* Wpk = (const float*)d_in[9];
  const float* Wpv = (const float*)d_in[10];
  const float* ptw = (const float*)d_in[11];
  const float* Wpb = (const float*)d_in[12];
  const float* bpb = (const float*)d_in[13];
  const float* Wo  = (const float*)d_in[14];
  const float* bo  = (const float*)d_in[15];
  const float* l1g = (const float*)d_in[16];
  const float* l1b = (const float*)d_in[17];
  const float* W1  = (const float*)d_in[18];
  const float* b1  = (const float*)d_in[19];
  const float* W2  = (const float*)d_in[20];
  const float* b2  = (const float*)d_in[21];
  const float* W3  = (const float*)d_in[22];
  const float* b3  = (const float*)d_in[23];
  const float* l2g = (const float*)d_in[24];
  const float* l2b = (const float*)d_in[25];
  float* out = (float*)d_out;

  char* cur = (char*)d_ws;
  auto alloc = [&](size_t bytes) -> void* {
    void* r = (void*)cur; cur += (bytes + 255) & ~(size_t)255; return r;
  };
  float*    raw    = (float*)   alloc((size_t)BN_ * QKV_ * 4);
  float*    qp     = (float*)   alloc((size_t)BN_ * 96 * 4);
  float*    kp     = (float*)   alloc((size_t)BN_ * 96 * 4);
  float*    vp     = (float*)   alloc((size_t)BN_ * 96 * 4);
  float*    cat    = (float*)   alloc((size_t)BN_ * CAT_ * 4);
  float*    x      = (float*)   alloc((size_t)BN_ * D_ * 4);
  float*    h1     = (float*)   alloc((size_t)BN_ * DH_ * 4);
  float*    h2     = (float*)   alloc((size_t)BN_ * DH_ * 4);
  float*    x3     = (float*)   alloc((size_t)BN_ * D_ * 4);
  _Float16* node_h = (_Float16*)alloc((size_t)BN_ * D_ * 2);
  _Float16* Wcat_h = (_Float16*)alloc((size_t)D_ * QKV_ * 2);
  _Float16* Wpb_h  = (_Float16*)alloc((size_t)E_ * H_ * 2);
  _Float16* cat_h  = (_Float16*)alloc((size_t)BN_ * CAT_ * 2);
  _Float16* Wo_h   = (_Float16*)alloc((size_t)CAT_ * D_ * 2);
  _Float16* x_h    = (_Float16*)alloc((size_t)BN_ * D_ * 2);
  _Float16* W1_h   = (_Float16*)alloc((size_t)D_ * DH_ * 2);
  _Float16* h1_h   = (_Float16*)alloc((size_t)BN_ * DH_ * 2);
  _Float16* W2_h   = (_Float16*)alloc((size_t)DH_ * DH_ * 2);
  _Float16* h2_h   = (_Float16*)alloc((size_t)BN_ * DH_ * 2);
  _Float16* W3_h   = (_Float16*)alloc((size_t)DH_ * D_ * 2);

  auto cvt = [&](const float* s, _Float16* d, int n) {
    cvt_kernel<<<(n + 255) / 256, 256, 0, stream>>>(s, d, n);
  };
  auto gemm = [&](const _Float16* A, const _Float16* Bm, const float* bias,
                  float* C, int M, int Nn, int K, int relu) {
    dim3 g(((Nn >> 4) + 3) / 4, M >> 4);
    gemm_f16_kernel<<<g, 128, 0, stream>>>(A, Bm, bias, C, M, Nn, K, relu);
  };

  // precision conversions
  cvt(node, node_h, BN_ * D_);
  pack_wcat_kernel<<<(D_ * QKV_ + 255) / 256, 256, 0, stream>>>(
      Wq, Wk, Wv, Wpq, Wpk, Wpv, Wcat_h, D_ * QKV_);
  cvt(Wpb, Wpb_h, E_ * H_);
  cvt(Wo, Wo_h, CAT_ * D_);
  cvt(W1, W1_h, D_ * DH_);
  cvt(W2, W2_h, DH_ * DH_);
  cvt(W3, W3_h, DH_ * D_);

  // projections (all q/k/v + point projections in one GEMM), then frames
  gemm(node_h, Wcat_h, nullptr, raw, BN_, QKV_, D_, 0);
  rotpts_kernel<<<(BN_ * 96 + 255) / 256, 256, 0, stream>>>(
      raw, rot, trans, qp, kp, vp, BN_ * 96);

  // fused IPA attention
  attn_kernel<<<dim3(N_, B_), 256, 0, stream>>>(
      raw, qp, kp, vp, edge, Wpb_h, ptw, bpb, rot, trans, cat);

  // output projection
  cvt(cat, cat_h, BN_ * CAT_);
  gemm(cat_h, Wo_h, bo, x, BN_, D_, CAT_, 0);

  // LN1 -> MLP -> LN2
  ln_kernel<<<BN_, 128, 0, stream>>>(x, l1g, l1b, nullptr, x_h, D_);
  gemm(x_h, W1_h, b1, h1, BN_, DH_, D_, 1);
  cvt(h1, h1_h, BN_ * DH_);
  gemm(h1_h, W2_h, b2, h2, BN_, DH_, DH_, 1);
  cvt(h2, h2_h, BN_ * DH_);
  gemm(h2_h, W3_h, b3, x3, BN_, D_, DH_, 0);
  ln_kernel<<<BN_, 128, 0, stream>>>(x3, l2g, l2b, out, nullptr, D_);
}
